// GroupedQueryAttention_85383949845167
// MI455X (gfx1250) — matrix operand order repaired and checked by reading, not compiled, not run on hardware
//
#include <hip/hip_runtime.h>

typedef _Float16 v16h __attribute__((ext_vector_type(16)));
typedef _Float16 v8h  __attribute__((ext_vector_type(8)));
typedef float    v8f  __attribute__((ext_vector_type(8)));
typedef float    v4f  __attribute__((ext_vector_type(4)));

// ---------------------------------------------------------------------------
// Address-space(1) load/store helpers: force GLOBAL_LOAD/STORE (not FLAT, which
// would also burn DScnt + the LDS pipe), with optional non-temporal hint for
// streaming tensors that must not evict the L2-resident working set.
// ---------------------------------------------------------------------------
template <typename V, bool NT>
__device__ __forceinline__ V gload(const void* p) {
  typedef const __attribute__((address_space(1))) V* gp_t;
  gp_t gp = (gp_t)(unsigned long long)p;
  if (NT) return __builtin_nontemporal_load(gp);
  return *gp;
}
template <typename V, bool NT>
__device__ __forceinline__ void gstore(void* p, V v) {
  typedef __attribute__((address_space(1))) V* gp_t;
  gp_t gp = (gp_t)(unsigned long long)p;
  if (NT) __builtin_nontemporal_store(v, gp);
  else *gp = v;
}

// ---------------------------------------------------------------------------
// Weight transpose + f32->f16 convert:  W[R][C] (row-major f32) -> WT[C][R] f16
// ---------------------------------------------------------------------------
__global__ __launch_bounds__(256)
void transpose_f32_to_f16(const float* __restrict__ W, _Float16* __restrict__ WT,
                          int R, int C) {
  __shared__ float tile[32][33];
  const int c0 = blockIdx.x * 32;
  const int r0 = blockIdx.y * 32;
  const int tx = threadIdx.x;   // 0..31
  const int ty = threadIdx.y;   // 0..7
#pragma unroll
  for (int i = 0; i < 4; ++i) {
    int r = ty + i * 8;
    tile[r][tx] = gload<float, false>(W + (size_t)(r0 + r) * C + (c0 + tx));
  }
  __syncthreads();
#pragma unroll
  for (int i = 0; i < 4; ++i) {
    int cc = ty + i * 8;
    gstore<_Float16, false>(WT + (size_t)(c0 + cc) * R + (r0 + tx),
                            (_Float16)tile[tx][cc]);
  }
}

// ---------------------------------------------------------------------------
// Generic WMMA GEMM:  C[z] = A[z] (MxK, row-major) * B[z]^T (B given as NxK
// row-major) * scale (+ bias).
//   MW:          rows per wave (16/32) -> MW/16 A-fragments reused across N
//   NT_:         16-col tiles per wave -> NT_ B-fragments reused across M
//   A_F32:       A is fp32, converted to f16 in-register while building frags
//   OUT_F16:     store C as f16 (intermediate tensors), else fp32
//   TRANS_STORE: store C transposed (used to produce V^T for the P*V GEMM)
//   HAS_BIAS:    compile-time bias enable (no runtime branch in epilogue)
//   A_NT:        non-temporal (last-use) A loads   (streaming attn matrix)
//   C_NT:        non-temporal C stores             (streaming scores / output)
// Batched over blockIdx.z: z -> (zb = z/ZH, zh = z%ZH); B-side head index is
// (zh >> gshift) to implement GQA's repeat_interleave.
// Block: 256 threads = 8 waves; block tile (8*MW) x (16*NT_).
// ---------------------------------------------------------------------------
template <int MW, int NT_, bool A_F32, bool OUT_F16, bool TRANS_STORE,
          bool HAS_BIAS, bool A_NT, bool C_NT>
__global__ __launch_bounds__(256)
void gemm_wmma(const void* __restrict__ Abase, int lda, long As_b, long As_h,
               const _Float16* __restrict__ Bbase, int ldb, long Bs_b, long Bs_g,
               int gshift,
               const float* __restrict__ bias,
               void* __restrict__ Cbase, int ldc, long Cs_b, long Cs_h,
               int K, int ZH, float scale) {
  constexpr int MT = MW / 16;  // A fragments / accumulator rows per wave
  const int z   = blockIdx.z;
  const int zb  = z / ZH;
  const int zh  = z - zb * ZH;
  const int tid  = threadIdx.x;
  const int wave = tid >> 5;
  const int lane = tid & 31;
  const int l15  = lane & 15;
  const int lhi  = lane >> 4;
  const int m0 = blockIdx.y * (8 * MW) + wave * MW;
  const int n0 = blockIdx.x * (16 * NT_);

  // ---- pointer-carried fragment addresses (bumped by +32 halves per step)
  const float*    ap32[MT];
  const _Float16* ap16[MT];
  if (A_F32) {
    const float* A32 = (const float*)Abase + (long)zb * As_b + (long)zh * As_h;
#pragma unroll
    for (int mi = 0; mi < MT; ++mi)
      ap32[mi] = A32 + (long)(m0 + mi * 16 + l15) * lda + lhi * 8;
  } else {
    const _Float16* A16 =
        (const _Float16*)Abase + (long)zb * As_b + (long)zh * As_h;
#pragma unroll
    for (int mi = 0; mi < MT; ++mi)
      ap16[mi] = A16 + (long)(m0 + mi * 16 + l15) * lda + lhi * 8;
  }
  const _Float16* Bz = Bbase + (long)zb * Bs_b + (long)(zh >> gshift) * Bs_g;
  const _Float16* bp[NT_];
#pragma unroll
  for (int j = 0; j < NT_; ++j)
    bp[j] = Bz + (long)(n0 + j * 16 + l15) * ldb + lhi * 8;

  v8f acc[MT][NT_];
#pragma unroll
  for (int mi = 0; mi < MT; ++mi)
#pragma unroll
    for (int j = 0; j < NT_; ++j)
#pragma unroll
      for (int r = 0; r < 8; ++r) acc[mi][j][r] = 0.0f;

  for (int ks = K >> 5; ks > 0; --ks) {
    // ---- A fragments: ISA 16-bit A 16x32 layout.
    // lane (lhi half): elems 0..7 = K[8*lhi .. +7], 8..15 = K[16+8*lhi .. +7]
    v16h a[MT];
#pragma unroll
    for (int mi = 0; mi < MT; ++mi) {
      if (A_F32) {
        v4f f0 = gload<v4f, A_NT>(ap32[mi]);
        v4f f1 = gload<v4f, A_NT>(ap32[mi] + 4);
        v4f f2 = gload<v4f, A_NT>(ap32[mi] + 16);
        v4f f3 = gload<v4f, A_NT>(ap32[mi] + 20);
        ap32[mi] += 32;
#pragma unroll
        for (int i = 0; i < 4; ++i) {
          a[mi][i]      = (_Float16)f0[i];
          a[mi][4 + i]  = (_Float16)f1[i];
          a[mi][8 + i]  = (_Float16)f2[i];
          a[mi][12 + i] = (_Float16)f3[i];
        }
      } else {
        v8h h0 = gload<v8h, A_NT>(ap16[mi]);
        v8h h1 = gload<v8h, A_NT>(ap16[mi] + 16);
        ap16[mi] += 32;
#pragma unroll
        for (int i = 0; i < 8; ++i) { a[mi][i] = h0[i]; a[mi][8 + i] = h1[i]; }
      }
    }

    // ---- NT_ B fragments, each reused by MT WMMAs. B layout: lane holds a
    // column, elems 0..7 = K[8*lhi .. +7], 8..15 = K[16+8*lhi .. +7] (B stored N x K row-major).
#pragma unroll
    for (int j = 0; j < NT_; ++j) {
      v8h b0 = gload<v8h, false>(bp[j]);
      v8h b1 = gload<v8h, false>(bp[j] + 16);
      bp[j] += 32;
      v16h bfrag;
#pragma unroll
      for (int i = 0; i < 8; ++i) { bfrag[i] = b0[i]; bfrag[8 + i] = b1[i]; }
#pragma unroll
      for (int mi = 0; mi < MT; ++mi)
        acc[mi][j] = __builtin_amdgcn_wmma_f32_16x16x32_f16(
            false, a[mi], false, bfrag, (short)0, acc[mi][j], false, false);
    }
  }

  // ---- Epilogue. C/D layout: VGPR r, lanes 0-15 -> M=r, lanes 16-31 -> M=r+8.
  float bval[NT_];
#pragma unroll
  for (int j = 0; j < NT_; ++j)
    bval[j] = HAS_BIAS ? bias[n0 + j * 16 + l15] : 0.0f;

  if (TRANS_STORE) {
    // one 64-bit column-base per j; per-row constant offsets
    if (OUT_F16) {
      _Float16* Cp = (_Float16*)Cbase + (long)zb * Cs_b + (long)zh * Cs_h;
#pragma unroll
      for (int j = 0; j < NT_; ++j) {
        _Float16* cc = Cp + (long)(n0 + j * 16 + l15) * ldc + m0 + 8 * lhi;
#pragma unroll
        for (int mi = 0; mi < MT; ++mi)
#pragma unroll
          for (int r = 0; r < 8; ++r) {
            float v = acc[mi][j][r] * scale;
            if (HAS_BIAS) v += bval[j];
            gstore<_Float16, C_NT>(cc + mi * 16 + r, (_Float16)v);
          }
      }
    } else {
      float* Cp = (float*)Cbase + (long)zb * Cs_b + (long)zh * Cs_h;
#pragma unroll
      for (int j = 0; j < NT_; ++j) {
        float* cc = Cp + (long)(n0 + j * 16 + l15) * ldc + m0 + 8 * lhi;
#pragma unroll
        for (int mi = 0; mi < MT; ++mi)
#pragma unroll
          for (int r = 0; r < 8; ++r) {
            float v = acc[mi][j][r] * scale;
            if (HAS_BIAS) v += bval[j];
            gstore<float, C_NT>(cc + mi * 16 + r, v);
          }
      }
    }
  } else {
    // one 64-bit row-base per (mi,r); per-column constant offsets (stride 16)
    if (OUT_F16) {
      _Float16* Cp = (_Float16*)Cbase + (long)zb * Cs_b + (long)zh * Cs_h;
#pragma unroll
      for (int mi = 0; mi < MT; ++mi)
#pragma unroll
        for (int r = 0; r < 8; ++r) {
          _Float16* rp =
              Cp + (long)(m0 + mi * 16 + r + 8 * lhi) * ldc + n0 + l15;
#pragma unroll
          for (int j = 0; j < NT_; ++j) {
            float v = acc[mi][j][r] * scale;
            if (HAS_BIAS) v += bval[j];
            gstore<_Float16, C_NT>(rp + j * 16, (_Float16)v);
          }
        }
    } else {
      float* Cp = (float*)Cbase + (long)zb * Cs_b + (long)zh * Cs_h;
#pragma unroll
      for (int mi = 0; mi < MT; ++mi)
#pragma unroll
        for (int r = 0; r < 8; ++r) {
          float* rp = Cp + (long)(m0 + mi * 16 + r + 8 * lhi) * ldc + n0 + l15;
#pragma unroll
          for (int j = 0; j < NT_; ++j) {
            float v = acc[mi][j][r] * scale;
            if (HAS_BIAS) v += bval[j];
            gstore<float, C_NT>(rp + j * 16, v);
          }
        }
    }
  }
}

// ---------------------------------------------------------------------------
// In-place row softmax over S=1024 fp32 values. grid = (S, B*H), block = 256.
// Fully streaming (268 MB >> L2): non-temporal loads and stores.
// ---------------------------------------------------------------------------
__global__ __launch_bounds__(256)
void softmax_rows(float* __restrict__ base) {
  float* row = base + ((size_t)blockIdx.y * 1024 + blockIdx.x) * 1024;
  const int tid  = threadIdx.x;
  const int wave = tid >> 5;
  const int lane = tid & 31;

  float v0 = gload<float, true>(row + tid);
  float v1 = gload<float, true>(row + tid + 256);
  float v2 = gload<float, true>(row + tid + 512);
  float v3 = gload<float, true>(row + tid + 768);
  float mx = fmaxf(fmaxf(v0, v1), fmaxf(v2, v3));
#pragma unroll
  for (int off = 16; off > 0; off >>= 1) mx = fmaxf(mx, __shfl_xor(mx, off, 32));

  __shared__ float red[8];
  if (lane == 0) red[wave] = mx;
  __syncthreads();
  float m = red[0];
#pragma unroll
  for (int i = 1; i < 8; ++i) m = fmaxf(m, red[i]);
  __syncthreads();

  float e0 = __expf(v0 - m), e1 = __expf(v1 - m);
  float e2 = __expf(v2 - m), e3 = __expf(v3 - m);
  float s = e0 + e1 + e2 + e3;
#pragma unroll
  for (int off = 16; off > 0; off >>= 1) s += __shfl_xor(s, off, 32);
  if (lane == 0) red[wave] = s;
  __syncthreads();
  float tot = 0.0f;
#pragma unroll
  for (int i = 0; i < 8; ++i) tot += red[i];
  const float inv = 1.0f / tot;

  gstore<float, true>(row + tid,       e0 * inv);
  gstore<float, true>(row + tid + 256, e1 * inv);
  gstore<float, true>(row + tid + 512, e2 * inv);
  gstore<float, true>(row + tid + 768, e3 * inv);
}

// ---------------------------------------------------------------------------
extern "C" void kernel_launch(void* const* d_in, const int* in_sizes, int n_in,
                              void* d_out, int out_size, void* d_ws, size_t ws_size,
                              hipStream_t stream) {
  (void)in_sizes; (void)n_in; (void)out_size; (void)ws_size;
  const int B = 4, S = 1024, D = 2048, H = 16, HD = 128, DK = 512;
  const float kScale = 0.08838834764831843f;  // 1/sqrt(128)

  const float* q_in = (const float*)d_in[0];
  const float* k_in = (const float*)d_in[1];
  const float* v_in = (const float*)d_in[2];
  const float* Wq   = (const float*)d_in[3];
  const float* bq   = (const float*)d_in[4];
  const float* Wk   = (const float*)d_in[5];
  const float* bk   = (const float*)d_in[6];
  const float* Wv   = (const float*)d_in[7];
  const float* bv   = (const float*)d_in[8];
  const float* Wo   = (const float*)d_in[9];
  const float* bo   = (const float*)d_in[10];

  // workspace layout (f16 intermediates, ~60 MB total)
  char*  ws  = (char*)d_ws;
  size_t off = 0;
  auto alloc = [&](size_t bytes) -> void* {
    void* p = ws + off;
    off += (bytes + 255) & ~(size_t)255;
    return p;
  };
  _Float16* WqT = (_Float16*)alloc((size_t)D * D * 2);       // [D(out) x D(in)]
  _Float16* WkT = (_Float16*)alloc((size_t)DK * D * 2);      // [DK x D]
  _Float16* WvT = (_Float16*)alloc((size_t)DK * D * 2);      // [DK x D]
  _Float16* WoT = (_Float16*)alloc((size_t)D * D * 2);       // [D x D]
  _Float16* Qh  = (_Float16*)alloc((size_t)B * S * D * 2);   // [B*S, D]
  _Float16* Kh  = (_Float16*)alloc((size_t)B * S * DK * 2);  // [B*S, DK]
  _Float16* Vt  = (_Float16*)alloc((size_t)DK * B * S * 2);  // [DK, B*S] (V^T)
  _Float16* Oh  = (_Float16*)alloc((size_t)B * S * D * 2);   // [B*S, D] attn out

  float* out  = (float*)d_out;                 // [B,S,D]
  float* attn = out + (size_t)B * S * D;       // [B,H,S,S]

  // 1) transpose+convert all weights to f16 NxK layout
  dim3 tb(32, 8);
  transpose_f32_to_f16<<<dim3(D / 32,  D / 32), tb, 0, stream>>>(Wq, WqT, D, D);
  transpose_f32_to_f16<<<dim3(DK / 32, D / 32), tb, 0, stream>>>(Wk, WkT, D, DK);
  transpose_f32_to_f16<<<dim3(DK / 32, D / 32), tb, 0, stream>>>(Wv, WvT, D, DK);
  transpose_f32_to_f16<<<dim3(D / 32,  D / 32), tb, 0, stream>>>(Wo, WoT, D, D);

  // 2) projections (A fp32 on-the-fly cvt, f16 out); 32x64 wave tiles
  gemm_wmma<32, 4, true, true, false, true, false, false>
      <<<dim3(D / 64, (B * S) / 256, 1), 256, 0, stream>>>(
      q_in, D, 0, 0, WqT, D, 0, 0, 0, bq, Qh, D, 0, 0, D, 1, 1.0f);
  gemm_wmma<32, 4, true, true, false, true, false, false>
      <<<dim3(DK / 64, (B * S) / 256, 1), 256, 0, stream>>>(
      k_in, D, 0, 0, WkT, D, 0, 0, 0, bk, Kh, DK, 0, 0, D, 1, 1.0f);
  // V projection stored transposed -> Vt[DK, B*S]
  gemm_wmma<32, 4, true, true, true, true, false, false>
      <<<dim3(DK / 64, (B * S) / 256, 1), 256, 0, stream>>>(
      v_in, D, 0, 0, WvT, D, 0, 0, 0, bv, Vt, B * S, 0, 0, D, 1, 1.0f);

  // 3) scores = Q K^T * 1/sqrt(hd), fp32 directly into attn_weights region.
  //    268 MB stream-out -> non-temporal stores (keep Qh/Kh resident in L2).
  //    z = b*H + h; B-side group index g = h >> 2
  gemm_wmma<32, 4, false, false, false, false, false, true>
      <<<dim3(S / 64, S / 256, B * H), 256, 0, stream>>>(
      Qh, D, (long)S * D, (long)HD,
      Kh, DK, (long)S * DK, (long)HD, 2,
      nullptr,
      attn, S, (long)H * S * S, (long)S * S,
      HD, H, kScale);

  // 4) softmax in place (this IS the attn_weights output), fully streaming
  softmax_rows<<<dim3(S, B * H), 256, 0, stream>>>(attn);

  // 5) attn_out = P V : A = fp32 attn (268 MB > L2, read exactly ONCE with
  //    non-temporal/last-use loads; full N=128 in one block-column, 16x128
  //    wave tiles). B = Vt slice (L2-resident). f16 out scattered into
  //    [B,S,H,hd] == [B*S, D].
  gemm_wmma<16, 8, true, true, false, false, true, false>
      <<<dim3(1, S / 128, B * H), 256, 0, stream>>>(
      attn, S, (long)H * S * S, (long)S * S,
      Vt, B * S, (long)S, (long)HD * B * S, 2,
      nullptr,
      Oh, D, (long)S * D, (long)HD,
      S, H, 1.0f);

  // 6) output = attn_out @ Wo + bo (fp32 out, never re-read -> NT stores)
  gemm_wmma<32, 4, false, false, false, true, false, true>
      <<<dim3(D / 64, (B * S) / 256, 1), 256, 0, stream>>>(
      Oh, D, 0, 0, WoT, D, 0, 0, 0, bo, out, D, 0, 0, D, 1, 1.0f);
}
